// GRU_Encoder_88545045774573
// MI455X (gfx1250) — matrix operand order repaired and checked by reading, not compiled, not run on hardware
//
#include <hip/hip_runtime.h>
#include <hip/hip_bf16.h>
#include <math.h>

typedef __bf16 bf16;
typedef __attribute__((ext_vector_type(16))) __bf16 v16bf;
typedef __attribute__((ext_vector_type(8)))  __bf16 v8bf;
typedef __attribute__((ext_vector_type(8)))  float  v8f;

#define EMBD 512
#define HID  1024
#define LATD 64
#define BATCH 64
#define SEQ  512

// workspace layout (bytes, all 256B aligned)
#define OFF_XBF   0ull          // [SEQ][BATCH][EMBD] bf16 : 33,554,432
#define OFF_WIHB  33554432ull   // [3H][EMBD] bf16        :  3,145,728
#define OFF_WHHB  36700160ull   // [3H][HID] bf16         :  6,291,456
#define OFF_HB0   42991616ull   // [BATCH][HID] bf16      :    131,072
#define OFF_HB1   43122688ull   // [BATCH][HID] bf16      :    131,072
#define OFF_HSUM  43253760ull   // [BATCH][HID] f32       :    262,144
#define OFF_HMBF  43515904ull   // [BATCH][HID] bf16      :    131,072
#define OFF_BAR   43646976ull   // 2 x u32 barrier

union V16U { v16bf v; v8bf h[2]; };

// A-matrix 16x32 bf16: lane<16 -> kb=0, lane>=16 -> kb=16;
// elems 0..15 = K kb..kb+15 (contiguous, matches B; two 16B loads)
__device__ __forceinline__ v16bf load_a_tile(const bf16* __restrict__ base, int ld, int k0, int lane) {
  const int m  = lane & 15;
  const int kb = (lane & 16) ? 16 : 0;
  const bf16* p = base + (size_t)m * ld + k0 + kb;
  V16U u;
  u.h[0] = *(const v8bf*)(p);
  u.h[1] = *(const v8bf*)(p + 8);
  return u.v;
}

// B-matrix 32x16 bf16: lane n=lane&15 is column (= weight row row0+n),
// lane<16 holds K k0..k0+15, lane>=16 holds K k0+16..k0+31 (contiguous)
__device__ __forceinline__ v16bf load_b_tile(const bf16* __restrict__ w, int ld, int row0, int k0, int lane) {
  const int n  = lane & 15;
  const int ko = (lane & 16) ? 16 : 0;
  const bf16* p = w + (size_t)(row0 + n) * ld + k0 + ko;
  V16U u;
  u.h[0] = *(const v8bf*)(p);
  u.h[1] = *(const v8bf*)(p + 8);
  return u.v;
}

// same B layout but converting fp32 source on the fly (small head matrices)
__device__ __forceinline__ v16bf load_b_tile_f32(const float* __restrict__ w, int ld, int row0, int k0, int lane) {
  const int n  = lane & 15;
  const int ko = (lane & 16) ? 16 : 0;
  const float* p = w + (size_t)(row0 + n) * ld + k0 + ko;
  V16U u;
#pragma unroll
  for (int e = 0; e < 16; ++e) u.v[e] = (bf16)p[e];
  return u.v;
}

__device__ __forceinline__ v8f wmma_bf16(v16bf a, v16bf b, v8f c) {
  return __builtin_amdgcn_wmma_f32_16x16x32_bf16(false, a, false, b, (short)0, c, false, false);
}

__device__ __forceinline__ void grid_sync(unsigned* bar, unsigned nblk) {
  __syncthreads();
  if (threadIdx.x == 0) {
    __threadfence();
    unsigned gen = __hip_atomic_load(bar + 1, __ATOMIC_RELAXED, __HIP_MEMORY_SCOPE_AGENT);
    unsigned arrived = __hip_atomic_fetch_add(bar, 1u, __ATOMIC_ACQ_REL, __HIP_MEMORY_SCOPE_AGENT);
    if (arrived == nblk - 1u) {
      __hip_atomic_store(bar, 0u, __ATOMIC_RELAXED, __HIP_MEMORY_SCOPE_AGENT);
      __hip_atomic_fetch_add(bar + 1, 1u, __ATOMIC_ACQ_REL, __HIP_MEMORY_SCOPE_AGENT);
    } else {
      while (__hip_atomic_load(bar + 1, __ATOMIC_ACQUIRE, __HIP_MEMORY_SCOPE_AGENT) == gen)
        __builtin_amdgcn_s_sleep(2);
    }
    __threadfence();
  }
  __syncthreads();
}

__global__ void prep_convert(const float* __restrict__ w_ih, const float* __restrict__ w_hh,
                             bf16* __restrict__ wihb, bf16* __restrict__ whhb,
                             bf16* __restrict__ hb0, unsigned* __restrict__ bar) {
  size_t i = (size_t)blockIdx.x * blockDim.x + threadIdx.x;
  size_t stride = (size_t)gridDim.x * blockDim.x;
  for (size_t k = i; k < (size_t)3 * HID * EMBD; k += stride) wihb[k] = (bf16)w_ih[k];
  for (size_t k = i; k < (size_t)3 * HID * HID;  k += stride) whhb[k] = (bf16)w_hh[k];
  for (size_t k = i; k < (size_t)BATCH * HID;    k += stride) hb0[k]  = (bf16)0.0f;
  if (i == 0) { bar[0] = 0u; bar[1] = 0u; }
}

__global__ void prep_gather(const int* __restrict__ ids, const float* __restrict__ emb,
                            bf16* __restrict__ xbf) {
  size_t i = (size_t)blockIdx.x * blockDim.x + threadIdx.x;
  size_t stride = (size_t)gridDim.x * blockDim.x;
  const size_t total = (size_t)SEQ * BATCH * EMBD;
  for (size_t f = i; f < total; f += stride) {
    int e   = (int)(f & (EMBD - 1));
    int row = (int)(f >> 9);          // row = t*BATCH + b
    int b   = row & (BATCH - 1);
    int t   = row >> 6;
    int id  = ids[b * SEQ + t];
    xbf[f] = (bf16)emb[(size_t)id * EMBD + e];
  }
}

// Persistent fused GRU: 32 blocks x 8 waves = 256 waves, one 16x16 h-tile per wave.
// All 8 waves of a block share one btile -> the A-tile (x rows + h rows) is staged
// once per block in LDS (48KB) and read by all waves via ds_load.
__global__ void __launch_bounds__(256, 1) gru_persistent(
    const float* __restrict__ b_ih, const float* __restrict__ b_hh,
    const bf16* __restrict__ xbf, const bf16* __restrict__ wihb, const bf16* __restrict__ whhb,
    bf16* __restrict__ hb0, bf16* __restrict__ hb1,
    float* __restrict__ hsum_out, unsigned* __restrict__ bar) {
  __shared__ __align__(16) bf16 Ash[16 * (EMBD + HID)];   // 48 KB
  bf16* Axs = Ash;                 // 16 x EMBD
  bf16* Ahs = Ash + 16 * EMBD;     // 16 x HID

  const int lane  = threadIdx.x & 31;
  const int wid   = blockIdx.x * (blockDim.x >> 5) + (threadIdx.x >> 5); // 0..255
  const int btile = wid >> 6;      // 0..3  (batch tile); uniform within a block
  const int jtile = wid & 63;      // 0..63 (hidden tile)
  const int j0    = jtile << 4;
  const int j     = j0 + (lane & 15);
  const int mh    = (lane & 16) ? 8 : 0;

  const float br_c = b_ih[j] + b_hh[j];               // r gate: biases fold
  const float bz_c = b_ih[HID + j] + b_hh[HID + j];   // z gate: biases fold
  const float bin  = b_ih[2 * HID + j];               // n gate: must stay split
  const float bhn  = b_hh[2 * HID + j];

  float hreg[8], hsum[8];
#pragma unroll
  for (int r = 0; r < 8; ++r) { hreg[r] = 0.0f; hsum[r] = 0.0f; }

  for (int t = 0; t < SEQ; ++t) {
    // ---- stage A-tile (16 x-rows, 16 h-rows) into LDS; both are contiguous ----
    const v8bf* xg = (const v8bf*)(xbf + (size_t)(t * BATCH + btile * 16) * EMBD);
    v8bf* xs = (v8bf*)Axs;
    for (int i = threadIdx.x; i < 16 * EMBD / 8; i += 256) xs[i] = xg[i];
    const v8bf* hg = (const v8bf*)((((t & 1) ? hb1 : hb0)) + (size_t)(btile * 16) * HID);
    v8bf* hs = (v8bf*)Ahs;
    for (int i = threadIdx.x; i < 16 * HID / 8; i += 256) hs[i] = hg[i];
    __syncthreads();

    v8f accr  = {0,0,0,0,0,0,0,0};
    v8f accz  = accr;
    v8f accnx = accr;   // x-side of n gate (r multiplies only h-side + b_hh)
    v8f accnh = accr;

    // x_t @ W_ih^T  (K = EMBD): A from LDS, B from L2-resident bf16 weights
#pragma unroll 2
    for (int k0 = 0; k0 < EMBD; k0 += 32) {
      v16bf a  = load_a_tile(Axs, EMBD, k0, lane);
      v16bf brt = load_b_tile(wihb, EMBD, j0,           k0, lane);
      v16bf bzt = load_b_tile(wihb, EMBD, HID + j0,     k0, lane);
      v16bf bnt = load_b_tile(wihb, EMBD, 2 * HID + j0, k0, lane);
      accr  = wmma_bf16(a, brt, accr);
      accz  = wmma_bf16(a, bzt, accz);
      accnx = wmma_bf16(a, bnt, accnx);
    }
    // h_{t-1} @ W_hh^T  (K = HID)
#pragma unroll 2
    for (int k0 = 0; k0 < HID; k0 += 32) {
      v16bf a  = load_a_tile(Ahs, HID, k0, lane);
      v16bf brt = load_b_tile(whhb, HID, j0,           k0, lane);
      v16bf bzt = load_b_tile(whhb, HID, HID + j0,     k0, lane);
      v16bf bnt = load_b_tile(whhb, HID, 2 * HID + j0, k0, lane);
      accr  = wmma_bf16(a, brt, accr);
      accz  = wmma_bf16(a, bzt, accz);
      accnh = wmma_bf16(a, bnt, accnh);
    }

    bf16* Hw = (t & 1) ? hb0 : hb1;
#pragma unroll
    for (int r = 0; r < 8; ++r) {
      float pr = accr[r] + br_c;
      float pz = accz[r] + bz_c;
      float rg = 1.0f / (1.0f + __expf(-pr));
      float zg = 1.0f / (1.0f + __expf(-pz));
      float nn = tanhf(accnx[r] + bin + rg * (accnh[r] + bhn));
      float hn = (1.0f - zg) * nn + zg * hreg[r];
      hreg[r] = hn;
      hsum[r] += hn;
      int brow = btile * 16 + mh + r;
      Hw[(size_t)brow * HID + j] = (bf16)hn;
    }
    // grid_sync's leading __syncthreads also fences this step's LDS reads
    // from next step's LDS refill.
    grid_sync(bar, gridDim.x);
  }

#pragma unroll
  for (int r = 0; r < 8; ++r) {
    int brow = btile * 16 + mh + r;
    hsum_out[(size_t)brow * HID + j] = hsum[r];
  }
}

// mean/logv heads: 1 block, 32 waves; wave -> one 16x16 output tile
__global__ void __launch_bounds__(1024, 1) vae_head(
    const float* __restrict__ hsum,
    const float* __restrict__ w_mean, const float* __restrict__ b_mean,
    const float* __restrict__ w_logv, const float* __restrict__ b_logv,
    bf16* __restrict__ hmbf, float* __restrict__ out) {
  for (int i = threadIdx.x; i < BATCH * HID; i += blockDim.x)
    hmbf[i] = (bf16)(hsum[i] * (1.0f / (float)SEQ));
  __syncthreads();

  const int lane  = threadIdx.x & 31;
  const int wv    = threadIdx.x >> 5;   // 0..31
  const int sel   = wv >> 4;            // 0 = mean, 1 = logv
  const int tid   = wv & 15;
  const int btile = tid >> 2;
  const int ltile = tid & 3;
  const float* W  = sel ? w_logv : w_mean;
  const float* Bv = sel ? b_logv : b_mean;

  v8f acc = {0,0,0,0,0,0,0,0};
  const bf16* A = hmbf + (size_t)(btile * 16) * HID;
#pragma unroll 2
  for (int k0 = 0; k0 < HID; k0 += 32) {
    v16bf a = load_a_tile(A, HID, k0, lane);
    v16bf b = load_b_tile_f32(W, HID, ltile * 16, k0, lane);
    acc = wmma_bf16(a, b, acc);
  }
  const int l = ltile * 16 + (lane & 15);
  const float bias = Bv[l];
  const int mh = (lane & 16) ? 8 : 0;
#pragma unroll
  for (int r = 0; r < 8; ++r) {
    int b = btile * 16 + mh + r;
    out[(size_t)sel * (BATCH * LATD) + b * LATD + l] = acc[r] + bias;
  }
}

extern "C" void kernel_launch(void* const* d_in, const int* in_sizes, int n_in,
                              void* d_out, int out_size, void* d_ws, size_t ws_size,
                              hipStream_t stream) {
  (void)in_sizes; (void)n_in; (void)out_size; (void)ws_size;
  const int*   ids    = (const int*)  d_in[0];
  const float* emb    = (const float*)d_in[1];
  const float* w_ih   = (const float*)d_in[2];
  const float* w_hh   = (const float*)d_in[3];
  const float* b_ih   = (const float*)d_in[4];
  const float* b_hh   = (const float*)d_in[5];
  const float* w_mean = (const float*)d_in[6];
  const float* b_mean = (const float*)d_in[7];
  const float* w_logv = (const float*)d_in[8];
  const float* b_logv = (const float*)d_in[9];

  char* ws = (char*)d_ws;
  bf16*     xbf  = (bf16*)(ws + OFF_XBF);
  bf16*     wihb = (bf16*)(ws + OFF_WIHB);
  bf16*     whhb = (bf16*)(ws + OFF_WHHB);
  bf16*     hb0  = (bf16*)(ws + OFF_HB0);
  bf16*     hb1  = (bf16*)(ws + OFF_HB1);
  float*    hsum = (float*)(ws + OFF_HSUM);
  bf16*     hmbf = (bf16*)(ws + OFF_HMBF);
  unsigned* bar  = (unsigned*)(ws + OFF_BAR);
  float*    out  = (float*)d_out;

  hipLaunchKernelGGL(prep_convert, dim3(512), dim3(256), 0, stream,
                     w_ih, w_hh, wihb, whhb, hb0, bar);
  hipLaunchKernelGGL(prep_gather, dim3(2048), dim3(256), 0, stream,
                     ids, emb, xbf);
  hipLaunchKernelGGL(gru_persistent, dim3(32), dim3(256), 0, stream,
                     b_ih, b_hh, xbf, wihb, whhb, hb0, hb1, hsum, bar);
  hipLaunchKernelGGL(vae_head, dim3(1), dim3(1024), 0, stream,
                     hsum, w_mean, b_mean, w_logv, b_logv, hmbf, out);
}
